// GCN3D_segR_30511447671642
// MI455X (gfx1250) — matrix operand order repaired and checked by reading, not compiled, not run on hardware
//
#include <hip/hip_runtime.h>
#include <hip/hip_bf16.h>
#include <math.h>

#define BATCH 4
#define V0 2048
#define V1N 512
#define V2N 128
#define NEI 10
#define SS 7
#define EPSBN 1e-5f
#define BIGF 3.0e38f

typedef __attribute__((ext_vector_type(16))) _Float16 v16h;
typedef __attribute__((ext_vector_type(8)))  float    v8f;

// ---------------------------------------------------------------------------
// kNN: per query point keep KP1 smallest distances (incl. self), emit KP1-1.
// Candidate coords staged through LDS in tiles of 256.
// ---------------------------------------------------------------------------
template<int KP1>
__global__ void knn_kernel(const float* __restrict__ verts, int* __restrict__ nidx,
                           int V, int nq, int qstride) {
  int b = blockIdx.y;
  int q = blockIdx.x * blockDim.x + threadIdx.x;
  bool active = (q < nq);
  int p = q * qstride;
  float px = 0.f, py = 0.f, pz = 0.f;
  if (active) {
    const float* pp = verts + ((size_t)b * V + p) * 3;
    px = pp[0]; py = pp[1]; pz = pp[2];
  }
  float bd[KP1]; int bi[KP1];
#pragma unroll
  for (int t = 0; t < KP1; ++t) { bd[t] = BIGF; bi[t] = 0; }
  __shared__ float sx[256], sy[256], sz[256];
  for (int t0 = 0; t0 < V; t0 += 256) {
    int lim = min(256, V - t0);
    for (int j = threadIdx.x; j < lim; j += blockDim.x) {
      const float* pj = verts + ((size_t)b * V + t0 + j) * 3;
      sx[j] = pj[0]; sy[j] = pj[1]; sz[j] = pj[2];
    }
    __syncthreads();
    if (active) {
      for (int j = 0; j < lim; ++j) {
        float dx = sx[j] - px, dy = sy[j] - py, dz = sz[j] - pz;
        float d = dx * dx + dy * dy + dz * dz;
        if (d < bd[KP1 - 1]) {
          int idxj = t0 + j;
          int pos = KP1 - 1;
#pragma unroll
          for (int t = KP1 - 1; t > 0; --t) {
            if (bd[t - 1] > d) { bd[t] = bd[t - 1]; bi[t] = bi[t - 1]; pos = t - 1; }
          }
          bd[pos] = d; bi[pos] = idxj;
        }
      }
    }
    __syncthreads();
  }
  if (active) {
    int* out = nidx + ((size_t)b * nq + q) * (KP1 - 1);
#pragma unroll
    for (int t = 1; t < KP1; ++t) out[t - 1] = bi[t];
  }
}

// ---------------------------------------------------------------------------
// conv_surface: fm0[b,v,c] (c<64) = relu( sum_s max_n relu(<ndn_n, sdn_{s,c}>) )
// grid (V0, B), block 64
// ---------------------------------------------------------------------------
__global__ void conv_surface_kernel(const float* __restrict__ verts,
                                    const int* __restrict__ nidx,
                                    const float* __restrict__ dir0,
                                    float* __restrict__ fm0) {
  int v = blockIdx.x, b = blockIdx.y;
  __shared__ float ndn[NEI][3];
  const float* pv = verts + ((size_t)b * V0 + v) * 3;
  if (threadIdx.x < NEI) {
    int j = nidx[((size_t)b * V0 + v) * NEI + threadIdx.x];
    const float* pj = verts + ((size_t)b * V0 + j) * 3;
    float dx = pj[0] - pv[0], dy = pj[1] - pv[1], dz = pj[2] - pv[2];
    float inv = 1.0f / fmaxf(sqrtf(dx * dx + dy * dy + dz * dz), 1e-12f);
    ndn[threadIdx.x][0] = dx * inv; ndn[threadIdx.x][1] = dy * inv; ndn[threadIdx.x][2] = dz * inv;
  }
  __syncthreads();
  int c = threadIdx.x;                 // 0..63
  const int W = SS * 64;               // 448 columns
  float acc = 0.f;
  for (int s = 0; s < SS; ++s) {
    int col = s * 64 + c;
    float d0 = dir0[col], d1 = dir0[W + col], d2 = dir0[2 * W + col];
    float inv = 1.0f / fmaxf(sqrtf(d0 * d0 + d1 * d1 + d2 * d2), 1e-12f);
    d0 *= inv; d1 *= inv; d2 *= inv;
    float m = -BIGF;
    for (int n = 0; n < NEI; ++n) {
      float th = fmaxf(ndn[n][0] * d0 + ndn[n][1] * d1 + ndn[n][2] * d2, 0.f);
      m = fmaxf(m, th);
    }
    acc += m;
  }
  fm0[((size_t)b * V0 + v) * 128 + c] = fmaxf(acc, 0.f);
}

// ---------------------------------------------------------------------------
// rgb linear + relu into fm0[..., 64:128].  grid (V0,B), block 64
// ---------------------------------------------------------------------------
__global__ void rgb_kernel(const float* __restrict__ rgb_f, const float* __restrict__ rgb_w,
                           const float* __restrict__ rgb_b, float* __restrict__ fm0) {
  int v = blockIdx.x, b = blockIdx.y, o = threadIdx.x;
  const float* rf = rgb_f + (size_t)b * 32 * V0 + v;
  float acc = rgb_b[o];
#pragma unroll
  for (int cc = 0; cc < 32; ++cc) acc += rf[(size_t)cc * V0] * rgb_w[o * 32 + cc];
  fm0[((size_t)b * V0 + v) * 128 + 64 + o] = fmaxf(acc, 0.f);
}

// ---------------------------------------------------------------------------
// per-channel sum / sumsq over rows (one block per channel, 256 threads)
// ---------------------------------------------------------------------------
__global__ void colstats_kernel(const float* __restrict__ x, int rows, int ld, int c0,
                                float* __restrict__ sums, float* __restrict__ sqs) {
  int c = blockIdx.x;
  float s = 0.f, qv = 0.f;
  for (int r = threadIdx.x; r < rows; r += blockDim.x) {
    float val = x[(size_t)r * ld + c0 + c];
    s += val; qv += val * val;
  }
  __shared__ float ss[256], qq[256];
  ss[threadIdx.x] = s; qq[threadIdx.x] = qv;
  __syncthreads();
  for (int o = 128; o > 0; o >>= 1) {
    if (threadIdx.x < o) { ss[threadIdx.x] += ss[threadIdx.x + o]; qq[threadIdx.x] += qq[threadIdx.x + o]; }
    __syncthreads();
  }
  if (threadIdx.x == 0) { sums[c] = ss[0]; sqs[c] = qq[0]; }
}

__global__ void bn_apply_kernel(float* __restrict__ x, int rows, int ld, int c0, int C,
                                const float* __restrict__ sums, const float* __restrict__ sqs,
                                const float* __restrict__ gamma, const float* __restrict__ beta,
                                int do_relu) {
  int idx = blockIdx.x * blockDim.x + threadIdx.x;
  if (idx >= rows * C) return;
  int r = idx / C, c = idx % C;
  float invn = 1.0f / (float)rows;
  float m = sums[c] * invn;
  float var = sqs[c] * invn - m * m;
  float v = x[(size_t)r * ld + c0 + c];
  v = gamma[c] * (v - m) * rsqrtf(var + EPSBN) + beta[c];
  if (do_relu) v = fmaxf(v, 0.f);
  x[(size_t)r * ld + c0 + c] = v;
}

// ---------------------------------------------------------------------------
// WMMA GEMM: C[M,N] = A[M,K] * W[K,N] + bias[N]   (f32 in, f16 WMMA, f32 acc)
// K and N are template constants -> K loop fully unrolled AND all B-load /
// C-store row strides become compile-time immediates (max K*N*4 = 4MB fits
// the signed 24-bit IOFFSET), eliminating per-row 64-bit address arithmetic.
// One wave computes a 16x32 output tile (two 16x16 WMMA tiles sharing one A
// fragment); block = 4 waves -> 128 output columns per block.
// A-fragment spans are 8 consecutive, 32B-aligned f32 -> float4 (b128) loads.
// B loads coalesce across each 16-lane half (consecutive n per lane).
// Fragment layouts per CDNA5 ISA 7.12.2 (wave32).
// ---------------------------------------------------------------------------
template<int K, int N>
__global__ void wmma_gemm_kernel(const float* __restrict__ A, const float* __restrict__ W,
                                 const float* __restrict__ bias, float* __restrict__ C) {
  int wave = threadIdx.x >> 5;
  int lane = threadIdx.x & 31;
  int tileM = blockIdx.y;
  int row  = lane & 15;
  int half = lane >> 4;
  int n0 = (blockIdx.x * 8 + wave * 2) * 16 + row;   // col for tile 0 (== store col)
  const float* Arow = A + (size_t)(tileM * 16 + row) * K;
  const float* Wc   = W + n0;                        // column base; rows via immediates
  v8f acc0 = {}, acc1 = {};
#pragma unroll
  for (int k0 = 0; k0 < K; k0 += 32) {
    int ka = k0 + half * 8;
    float af[16];
    *(float4*)(af + 0)  = *(const float4*)(Arow + ka);
    *(float4*)(af + 4)  = *(const float4*)(Arow + ka + 4);
    *(float4*)(af + 8)  = *(const float4*)(Arow + ka + 16);
    *(float4*)(af + 12) = *(const float4*)(Arow + ka + 20);
    v16h a;
#pragma unroll
    for (int i = 0; i < 16; ++i) a[i] = (_Float16)af[i];
    const float* Wk = Wc + (size_t)half * 8 * N;     // only runtime row component
    v16h b0, b1;
#pragma unroll
    for (int j = 0; j < 8; ++j) {
      b0[2 * j]     = (_Float16)Wk[(size_t)(k0 + 2 * j + (j & 4) * 2) * N];
      b0[2 * j + 1] = (_Float16)Wk[(size_t)(k0 + 2 * j + 1 + (j & 4) * 2) * N];
      b1[2 * j]     = (_Float16)Wk[(size_t)(k0 + 2 * j + (j & 4) * 2) * N + 16];
      b1[2 * j + 1] = (_Float16)Wk[(size_t)(k0 + 2 * j + 1 + (j & 4) * 2) * N + 16];
    }
    acc0 = __builtin_amdgcn_wmma_f32_16x16x32_f16(false, a, false, b0,
                                                  (short)0, acc0, false, false);
    acc1 = __builtin_amdgcn_wmma_f32_16x16x32_f16(false, a, false, b1,
                                                  (short)0, acc1, false, false);
  }
  float bv0 = bias[n0];
  float bv1 = bias[n0 + 16];
  float* Cc = C + (size_t)(tileM * 16 + 8 * half) * N + n0;  // rows via immediates
#pragma unroll
  for (int r = 0; r < 8; ++r) {
    Cc[(size_t)r * N]      = acc0[r] + bv0;
    Cc[(size_t)r * N + 16] = acc1[r] + bv1;
  }
}

// ---------------------------------------------------------------------------
// conv_layer aggregation: fm = f_center + sum_s max_n relu(<ndn,sdn>)*f_support
// grid (V, B), blockDim.x == out_c
// ---------------------------------------------------------------------------
__global__ void conv_agg_kernel(const float* __restrict__ f_out, const int* __restrict__ nidx,
                                const float* __restrict__ verts, const float* __restrict__ dir,
                                float* __restrict__ fm, int V, int out_c) {
  int v = blockIdx.x, b = blockIdx.y;
  __shared__ float ndn[NEI][3];
  __shared__ int sidx[NEI];
  const float* pv = verts + ((size_t)b * V + v) * 3;
  if (threadIdx.x < NEI) {
    int j = nidx[((size_t)b * V + v) * NEI + threadIdx.x];
    sidx[threadIdx.x] = j;
    const float* pj = verts + ((size_t)b * V + j) * 3;
    float dx = pj[0] - pv[0], dy = pj[1] - pv[1], dz = pj[2] - pv[2];
    float inv = 1.0f / fmaxf(sqrtf(dx * dx + dy * dy + dz * dz), 1e-12f);
    ndn[threadIdx.x][0] = dx * inv; ndn[threadIdx.x][1] = dy * inv; ndn[threadIdx.x][2] = dz * inv;
  }
  __syncthreads();
  int c = threadIdx.x;
  int F = (SS + 1) * out_c;
  int sdc = SS * out_c;
  float acc = f_out[((size_t)b * V + v) * F + c];
  for (int s = 0; s < SS; ++s) {
    int col = s * out_c + c;
    float d0 = dir[col], d1 = dir[sdc + col], d2 = dir[2 * sdc + col];
    float inv = 1.0f / fmaxf(sqrtf(d0 * d0 + d1 * d1 + d2 * d2), 1e-12f);
    d0 *= inv; d1 *= inv; d2 *= inv;
    float m = -BIGF;
    for (int n = 0; n < NEI; ++n) {
      float th = fmaxf(ndn[n][0] * d0 + ndn[n][1] * d1 + ndn[n][2] * d2, 0.f);
      float sup = f_out[((size_t)b * V + sidx[n]) * F + out_c + col];
      m = fmaxf(m, th * sup);
    }
    acc += m;
  }
  fm[((size_t)b * V + v) * out_c + c] = acc;
}

// ---------------------------------------------------------------------------
// pool: max over 4 neighbors of sampled points + copy sampled vertices
// grid (pnum, B)
// ---------------------------------------------------------------------------
__global__ void pool_sample_kernel(const float* __restrict__ verts, const float* __restrict__ fm,
                                   const int* __restrict__ nidx4, float* __restrict__ v_out,
                                   float* __restrict__ fm_out, int V, int pnum, int C) {
  int i = blockIdx.x, b = blockIdx.y;
  const int* nb = nidx4 + ((size_t)b * pnum + i) * 4;
  int n0 = nb[0], n1 = nb[1], n2 = nb[2], n3 = nb[3];
  for (int c = threadIdx.x; c < C; c += blockDim.x) {
    float m = fm[((size_t)b * V + n0) * C + c];
    m = fmaxf(m, fm[((size_t)b * V + n1) * C + c]);
    m = fmaxf(m, fm[((size_t)b * V + n2) * C + c]);
    m = fmaxf(m, fm[((size_t)b * V + n3) * C + c]);
    fm_out[((size_t)b * pnum + i) * C + c] = m;
  }
  if (threadIdx.x < 3)
    v_out[((size_t)b * pnum + i) * 3 + threadIdx.x] = verts[((size_t)b * V + i * 4) * 3 + threadIdx.x];
}

// ---------------------------------------------------------------------------
// nearest source index (argmin, first-hit ties like jnp.argmin)
// ---------------------------------------------------------------------------
__global__ void nearest_kernel(const float* __restrict__ tgt, const float* __restrict__ src,
                               int* __restrict__ out, int Vt, int Vs) {
  __shared__ float s[V1N * 3];
  int b = blockIdx.y;
  for (int j = threadIdx.x; j < Vs * 3; j += blockDim.x) s[j] = src[(size_t)b * Vs * 3 + j];
  __syncthreads();
  int i = blockIdx.x * blockDim.x + threadIdx.x;
  if (i >= Vt) return;
  const float* pt = tgt + ((size_t)b * Vt + i) * 3;
  float px = pt[0], py = pt[1], pz = pt[2];
  float best = BIGF; int bi = 0;
  for (int j = 0; j < Vs; ++j) {
    float dx = s[j * 3] - px, dy = s[j * 3 + 1] - py, dz = s[j * 3 + 2] - pz;
    float d = dx * dx + dy * dy + dz * dz;
    if (d < best) { best = d; bi = j; }
  }
  out[(size_t)b * Vt + i] = bi;
}

// global max over the 128 coarse points, per (b, c)
__global__ void fglobal_kernel(const float* __restrict__ fm4, float* __restrict__ fg) {
  int b = blockIdx.x, c = threadIdx.x;   // 512 threads
  float m = -BIGF;
  for (int r = 0; r < V2N; ++r) m = fmaxf(m, fm4[((size_t)b * V2N + r) * 512 + c]);
  fg[b * 512 + c] = m;
}

// ---------------------------------------------------------------------------
// final channel-concat + transpose:  out1 = feat (B,1280,V), out2 = fuse (B,1792,V)
// grid (V0/256, B), block 256 (threads along v -> coalesced stores)
// ---------------------------------------------------------------------------
__global__ void assemble_kernel(const float* __restrict__ fm0, const float* __restrict__ fm1,
                                const float* __restrict__ fm2, const float* __restrict__ fm3,
                                const float* __restrict__ fm4, const float* __restrict__ fglob,
                                const int* __restrict__ near1, const int* __restrict__ near2,
                                float* __restrict__ out1, float* __restrict__ out2) {
  int b = blockIdx.y;
  int v = blockIdx.x * blockDim.x + threadIdx.x;
  int n1 = near1[(size_t)b * V0 + v];
  int n2 = near2[(size_t)b * V0 + v];
  size_t o1 = (size_t)b * 1280 * V0, o2 = (size_t)b * 1792 * V0;
  for (int c = 0; c < 1792; ++c) {
    float val;
    if (c < 128)       val = fm0[((size_t)b * V0 + v) * 128 + c];
    else if (c < 256)  val = fm1[((size_t)b * V0 + v) * 128 + (c - 128)];
    else if (c < 512)  val = fm2[((size_t)b * V1N + n1) * 256 + (c - 256)];
    else if (c < 768)  val = fm3[((size_t)b * V1N + n1) * 256 + (c - 512)];
    else if (c < 1280) val = fm4[((size_t)b * V2N + n2) * 512 + (c - 768)];
    else               val = fglob[b * 512 + (c - 1280)];
    out2[o2 + (size_t)c * V0 + v] = val;
    if (c < 1280) out1[o1 + (size_t)c * V0 + v] = val;
  }
}

// ---------------------------------------------------------------------------
extern "C" void kernel_launch(void* const* d_in, const int* in_sizes, int n_in,
                              void* d_out, int out_size, void* d_ws, size_t ws_size,
                              hipStream_t stream) {
  const float* vertices = (const float*)d_in[0];
  const float* rgb_f    = (const float*)d_in[1];
  const float* dir0     = (const float*)d_in[2];
  const float* w1       = (const float*)d_in[3];
  const float* b1       = (const float*)d_in[4];
  const float* dir1     = (const float*)d_in[5];
  const float* w2       = (const float*)d_in[6];
  const float* b2       = (const float*)d_in[7];
  const float* dir2     = (const float*)d_in[8];
  const float* w3       = (const float*)d_in[9];
  const float* b3       = (const float*)d_in[10];
  const float* dir3     = (const float*)d_in[11];
  const float* w4       = (const float*)d_in[12];
  const float* b4       = (const float*)d_in[13];
  const float* dir4     = (const float*)d_in[14];
  const float* rgb_w    = (const float*)d_in[15];
  const float* rgb_b    = (const float*)d_in[16];
  const float* rgb_bn_g = (const float*)d_in[17];
  const float* rgb_bn_b = (const float*)d_in[18];
  const float* bn1_g    = (const float*)d_in[19];
  const float* bn1_b    = (const float*)d_in[20];
  const float* bn2_g    = (const float*)d_in[21];
  const float* bn2_b    = (const float*)d_in[22];
  const float* bn3_g    = (const float*)d_in[23];
  const float* bn3_b    = (const float*)d_in[24];

  // workspace layout (float offsets)
  float* fws      = (float*)d_ws;
  float* gemm_out = fws;                       // 8,388,608
  float* fm0      = fws + 8388608;             // 1,048,576
  float* fm1      = fws + 9437184;             // 1,048,576
  float* fmp1     = fws + 10485760;            //   262,144
  float* fm2      = fws + 10747904;            //   524,288
  float* fm3      = fws + 11272192;            //   524,288
  float* fmp2     = fws + 11796480;            //   131,072
  float* fm4      = fws + 11927552;            //   262,144
  float* fglob    = fws + 12189696;            //     2,048
  float* v1       = fws + 12191744;            //     6,144
  float* v2       = fws + 12197888;            //     1,536
  float* sums     = fws + 12199424;            //       512
  float* sqs      = fws + 12199936;            //       512
  int*   iws      = (int*)(fws + 12200448);
  int*   nidx0    = iws;                       //  81,920
  int*   nidxp1   = iws + 81920;               //   8,192
  int*   nidx1    = iws + 90112;               //  20,480
  int*   nidxp2   = iws + 110592;              //   2,048
  int*   nidx2    = iws + 112640;              //   5,120
  int*   near1    = iws + 117760;              //   8,192
  int*   near2    = iws + 125952;              //   8,192

  // Stage 0: kNN(10) on full cloud
  knn_kernel<11><<<dim3(16, BATCH), 128, 0, stream>>>(vertices, nidx0, V0, V0, 1);

  // fm_0 = [conv_surface | bn(relu(rgb linear))]
  conv_surface_kernel<<<dim3(V0, BATCH), 64, 0, stream>>>(vertices, nidx0, dir0, fm0);
  rgb_kernel<<<dim3(V0, BATCH), 64, 0, stream>>>(rgb_f, rgb_w, rgb_b, fm0);
  colstats_kernel<<<64, 256, 0, stream>>>(fm0, BATCH * V0, 128, 64, sums, sqs);
  bn_apply_kernel<<<(BATCH * V0 * 64 + 255) / 256, 256, 0, stream>>>(
      fm0, BATCH * V0, 128, 64, 64, sums, sqs, rgb_bn_g, rgb_bn_b, 0);

  // conv_layer 1: (8192x128)@(128x1024)
  wmma_gemm_kernel<128, 1024><<<dim3(1024 / 128, (BATCH * V0) / 16), 128, 0, stream>>>(
      fm0, w1, b1, gemm_out);
  conv_agg_kernel<<<dim3(V0, BATCH), 128, 0, stream>>>(gemm_out, nidx0, vertices, dir1, fm1, V0, 128);
  colstats_kernel<<<128, 256, 0, stream>>>(fm1, BATCH * V0, 128, 0, sums, sqs);
  bn_apply_kernel<<<(BATCH * V0 * 128 + 255) / 256, 256, 0, stream>>>(
      fm1, BATCH * V0, 128, 0, 128, sums, sqs, bn1_g, bn1_b, 1);

  // pool 1 (rate 4, k=4) -> v1 (512 pts), fmp1
  knn_kernel<5><<<dim3(4, BATCH), 128, 0, stream>>>(vertices, nidxp1, V0, V1N, 4);
  pool_sample_kernel<<<dim3(V1N, BATCH), 128, 0, stream>>>(vertices, fm1, nidxp1, v1, fmp1, V0, V1N, 128);
  knn_kernel<11><<<dim3(4, BATCH), 128, 0, stream>>>(v1, nidx1, V1N, V1N, 1);

  // conv_layer 2: (2048x128)@(128x2048)
  wmma_gemm_kernel<128, 2048><<<dim3(2048 / 128, (BATCH * V1N) / 16), 128, 0, stream>>>(
      fmp1, w2, b2, gemm_out);
  conv_agg_kernel<<<dim3(V1N, BATCH), 256, 0, stream>>>(gemm_out, nidx1, v1, dir2, fm2, V1N, 256);
  colstats_kernel<<<256, 256, 0, stream>>>(fm2, BATCH * V1N, 256, 0, sums, sqs);
  bn_apply_kernel<<<(BATCH * V1N * 256 + 255) / 256, 256, 0, stream>>>(
      fm2, BATCH * V1N, 256, 0, 256, sums, sqs, bn2_g, bn2_b, 1);

  // conv_layer 3: (2048x256)@(256x2048)
  wmma_gemm_kernel<256, 2048><<<dim3(2048 / 128, (BATCH * V1N) / 16), 128, 0, stream>>>(
      fm2, w3, b3, gemm_out);
  conv_agg_kernel<<<dim3(V1N, BATCH), 256, 0, stream>>>(gemm_out, nidx1, v1, dir3, fm3, V1N, 256);
  colstats_kernel<<<256, 256, 0, stream>>>(fm3, BATCH * V1N, 256, 0, sums, sqs);
  bn_apply_kernel<<<(BATCH * V1N * 256 + 255) / 256, 256, 0, stream>>>(
      fm3, BATCH * V1N, 256, 0, 256, sums, sqs, bn3_g, bn3_b, 1);

  // pool 2 -> v2 (128 pts), fmp2
  knn_kernel<5><<<dim3(1, BATCH), 128, 0, stream>>>(v1, nidxp2, V1N, V2N, 4);
  pool_sample_kernel<<<dim3(V2N, BATCH), 256, 0, stream>>>(v1, fm3, nidxp2, v2, fmp2, V1N, V2N, 256);
  knn_kernel<11><<<dim3(1, BATCH), 128, 0, stream>>>(v2, nidx2, V2N, V2N, 1);

  // conv_layer 4: (512x256)@(256x4096), no bn
  wmma_gemm_kernel<256, 4096><<<dim3(4096 / 128, (BATCH * V2N) / 16), 128, 0, stream>>>(
      fmp2, w4, b4, gemm_out);
  conv_agg_kernel<<<dim3(V2N, BATCH), 512, 0, stream>>>(gemm_out, nidx2, v2, dir4, fm4, V2N, 512);
  fglobal_kernel<<<BATCH, 512, 0, stream>>>(fm4, fglob);

  // nearest-neighbor upsample indices
  nearest_kernel<<<dim3(V0 / 256, BATCH), 256, 0, stream>>>(vertices, v1, near1, V0, V1N);
  nearest_kernel<<<dim3(V0 / 256, BATCH), 256, 0, stream>>>(vertices, v2, near2, V0, V2N);

  // final concat + transpose
  float* out1 = (float*)d_out;
  float* out2 = out1 + (size_t)BATCH * 1280 * V0;
  assemble_kernel<<<dim3(V0 / 256, BATCH), 256, 0, stream>>>(
      fm0, fm1, fm2, fm3, fm4, fglob, near1, near2, out1, out2);
}
